// MapEncoder_78426102825265
// MI455X (gfx1250) — matrix operand order repaired and checked by reading, not compiled, not run on hardware
//
#include <hip/hip_runtime.h>
#include <hip/hip_bf16.h>
#include <stddef.h>

// ---------------------------------------------------------------------------
// MapEncoder for MI455X (gfx1250): all 128-deep GEMMs via v_wmma_f32_16x16x32_f16
// (f16 A/B, f32 accumulate). Memory-bound workload -> f16 activations for GEMM
// inputs, f32 elsewhere. GEMM register-blocks 8 N-tiles per wave (16x128 strip)
// with K templated (128/512) so all tile addresses fold into instruction
// immediate offsets: no per-iteration VALU address math -> no WMMA->VALU hazard
// NOPs, loads issue in clauses with partial waits. Segment softmax via
// orderable-uint atomicMax + f32 atomicAdd. wave32 throughout.
// ---------------------------------------------------------------------------

#define NC_  50000
#define NL_  20000
#define EC_  200000
#define EL_  300000
#define HD_  128
#define EPS_ 1e-5f
#define PI_  3.14159265358979323846f

typedef __attribute__((ext_vector_type(16))) _Float16 v16h;
typedef __attribute__((ext_vector_type(8)))  _Float16 v8h;
typedef __attribute__((ext_vector_type(8)))  float    v8f;

// ------------------------------- device utils ------------------------------

__device__ __forceinline__ float warp_sum32(float v) {
  v += __shfl_xor(v, 16, 32);
  v += __shfl_xor(v, 8, 32);
  v += __shfl_xor(v, 4, 32);
  v += __shfl_xor(v, 2, 32);
  v += __shfl_xor(v, 1, 32);
  return v;
}

__device__ __forceinline__ unsigned fenc(float f) {
  unsigned u = __float_as_uint(f);
  return (u >> 31) ? ~u : (u | 0x80000000u);
}
__device__ __forceinline__ float fdec(unsigned u) {
  unsigned v = (u >> 31) ? (u & 0x7fffffffu) : ~u;
  return __uint_as_float(v);
}

__device__ __forceinline__ float wrap_angle(float a) {
  float r = fmodf(a + PI_, 2.0f * PI_);
  if (r < 0.0f) r += 2.0f * PI_;
  return r - PI_;
}

// ------------------------- WMMA GEMM (core kernel) --------------------------
// C[M,N] (f32 and/or f16) = act( A[M,KC](f16) @ Bt[N,KC]^T(f16) + bias[N] )
// One wave computes a 16(M) x 128(N) strip: 8 accumulators, A fragment loaded
// once per K-step and reused across the 8 N-tiles. KC is a template constant
// (128 or 512) so every load in the unrolled 128-K chunk is base+immediate.
// Fragment layouts per CDNA5 ISA 7.12.2:
//   A 16x32 f16 : lane<16 -> M=lane,    K-halfgroups at {0..7,16..23}
//                 lane>=16 -> M=lane-16, K-halfgroups at {8..15,24..31}
//   B 32x16 f16 : lane<16 -> N=lane, K at {0..7,16..23} ; lane>=16 -> N=lane-16, K at {8..15,24..31}
//   C 16x16 f32 : acc[r] -> M = r + (lane<16?0:8), N = lane&15
#define GEMM_WAVES 4
#define GEMM_NT    8   // N-tiles per wave (128 columns)

template <int KC>
__global__ void k_wmma_gemm_t(const _Float16* __restrict__ A,
                              const _Float16* __restrict__ Bt,
                              const float* __restrict__ bias,
                              float* __restrict__ Cf,
                              _Float16* __restrict__ Ch,
                              int M, int N, int relu) {
  const int lane = threadIdx.x & 31;
  const int wave = threadIdx.x >> 5;
  const int mtile = blockIdx.x * GEMM_WAVES + wave;
  if (mtile * 16 >= M) return;  // wave-uniform exit

  const int m = lane & 15;
  int row = mtile * 16 + m;
  if (row >= M) row = M - 1;                 // clamp loads on the M tail
  const int koff  = (lane >> 4) * 8;         // A half-group offset
  const int kboff = (lane >> 4) * 8;         // B half-group offset
  const int ncol0 = blockIdx.y * (GEMM_NT * 16) + (lane & 15);

  const _Float16* Ap = A  + (size_t)row * KC + koff;
  const _Float16* Bp = Bt + (size_t)ncol0 * KC + kboff;

  v8f acc[GEMM_NT] = {};
  for (int kb = 0; kb < KC / 128; ++kb) {    // one pointer bump per 128-K chunk
#pragma unroll
    for (int ks = 0; ks < 4; ++ks) {         // k0 = ks*32, immediate offsets
      v8h a0 = *(const v8h*)(Ap + ks * 32);
      v8h a1 = *(const v8h*)(Ap + ks * 32 + 16);
      const v16h av = __builtin_shufflevector(a0, a1,
          0,1,2,3,4,5,6,7,8,9,10,11,12,13,14,15);
#pragma unroll
      for (int nt = 0; nt < GEMM_NT; ++nt) {
        const _Float16* bp = Bp + nt * (16 * KC) + ks * 32;  // immediate offset
        v8h b0 = *(const v8h*)(bp);
        v8h b1 = *(const v8h*)(bp + 16);
        const v16h bv = __builtin_shufflevector(b0, b1,
            0,1,2,3,4,5,6,7,8,9,10,11,12,13,14,15);
        acc[nt] = __builtin_amdgcn_wmma_f32_16x16x32_f16(false, av, false, bv,
                                                         (short)0, acc[nt], false, false);
      }
    }
    Ap += 128;
    Bp += 128;
  }

#pragma unroll
  for (int nt = 0; nt < GEMM_NT; ++nt) {
    const int n  = blockIdx.y * (GEMM_NT * 16) + nt * 16 + (lane & 15);
    const float bc = bias ? bias[n] : 0.0f;
#pragma unroll
    for (int r = 0; r < 8; ++r) {
      const int mm   = (lane < 16) ? r : (r + 8);
      const int rrow = mtile * 16 + mm;
      if (rrow < M) {
        float v = acc[nt][r] + bc;
        if (relu) v = fmaxf(v, 0.0f);
        if (Cf) Cf[(size_t)rrow * N + n] = v;
        if (Ch) Ch[(size_t)rrow * N + n] = (_Float16)v;
      }
    }
  }
}

// ------------------------- weight convert/transpose -------------------------
// Bt[n*K + k] = (f16) W[k*N + n]
__global__ void k_w_to_f16t(const float* __restrict__ W, _Float16* __restrict__ Bt,
                            int K, int N) {
  int i = blockIdx.x * blockDim.x + threadIdx.x;
  if (i >= K * N) return;
  int k = i / N, n = i - k * N;
  Bt[(size_t)n * K + k] = (_Float16)W[i];
}

// -------------------- front linear (din<=8) + LN + ReLU ---------------------
// H[row,128] = relu(LN(X[row,:din] @ W1 + b1) * g + bln)   (f16 out)
__global__ void k_front(const float* __restrict__ X, int din, int nrows,
                        const float* __restrict__ W1, const float* __restrict__ b1,
                        const float* __restrict__ g, const float* __restrict__ bln,
                        _Float16* __restrict__ Hout) {
  const int lane = threadIdx.x & 31;
  const int wave = threadIdx.x >> 5;
  const int row = blockIdx.x * (blockDim.x >> 5) + wave;
  if (row >= nrows) return;
  float x[8];
  for (int d = 0; d < din; ++d) x[d] = X[(size_t)row * din + d];
  float acc[4];
#pragma unroll
  for (int t = 0; t < 4; ++t) {
    const int j = lane * 4 + t;
    float a = b1[j];
    for (int d = 0; d < din; ++d) a += x[d] * W1[d * HD_ + j];
    acc[t] = a;
  }
  float s = warp_sum32(acc[0] + acc[1] + acc[2] + acc[3]);
  const float mu = s * (1.0f / HD_);
  float vs = 0.0f;
#pragma unroll
  for (int t = 0; t < 4; ++t) { float d = acc[t] - mu; vs += d * d; }
  vs = warp_sum32(vs);
  const float rin = rsqrtf(vs * (1.0f / HD_) + EPS_);
#pragma unroll
  for (int t = 0; t < 4; ++t) {
    const int j = lane * 4 + t;
    float y = (acc[t] - mu) * rin * g[j] + bln[j];
    y = fmaxf(y, 0.0f);
    Hout[(size_t)row * HD_ + j] = (_Float16)y;
  }
}

// ------------------------------ LayerNorm(128) ------------------------------
__global__ void k_ln128(const float* __restrict__ X,
                        const float* __restrict__ g, const float* __restrict__ b,
                        int nrows, float* __restrict__ Yf, _Float16* __restrict__ Yh) {
  const int lane = threadIdx.x & 31;
  const int wave = threadIdx.x >> 5;
  const int row = blockIdx.x * (blockDim.x >> 5) + wave;
  if (row >= nrows) return;
  float v[4];
#pragma unroll
  for (int t = 0; t < 4; ++t) v[t] = X[(size_t)row * HD_ + lane * 4 + t];
  float s = warp_sum32(v[0] + v[1] + v[2] + v[3]);
  const float mu = s * (1.0f / HD_);
  float vs = 0.0f;
#pragma unroll
  for (int t = 0; t < 4; ++t) { float d = v[t] - mu; vs += d * d; }
  vs = warp_sum32(vs);
  const float rin = rsqrtf(vs * (1.0f / HD_) + EPS_);
#pragma unroll
  for (int t = 0; t < 4; ++t) {
    const int j = lane * 4 + t;
    const float y = (v[t] - mu) * rin * g[j] + b[j];
    if (Yf) Yf[(size_t)row * HD_ + j] = y;
    if (Yh) Yh[(size_t)row * HD_ + j] = (_Float16)y;
  }
}

// ----------------------------- feature builders -----------------------------
__global__ void k_pack_lfeat(const float* __restrict__ len, const float* __restrict__ isect,
                             const float* __restrict__ turn, const float* __restrict__ tc,
                             float* __restrict__ F, int n) {
  int i = blockIdx.x * blockDim.x + threadIdx.x;
  if (i >= n) return;
  F[i * 4 + 0] = len[i];
  F[i * 4 + 1] = isect[i];
  F[i * 4 + 2] = turn[i];
  F[i * 4 + 3] = tc[i];
}

__global__ void k_c2l_feat(const float* __restrict__ cpos, const float* __restrict__ lpos,
                           const float* __restrict__ chead, const float* __restrict__ lhead,
                           const int* __restrict__ src, const int* __restrict__ dst,
                           float* __restrict__ F, int E) {
  int i = blockIdx.x * blockDim.x + threadIdx.x;
  if (i >= E) return;
  const int s = src[i], d = dst[i];
  const float rx = cpos[2 * s] - lpos[2 * d];
  const float ry = cpos[2 * s + 1] - lpos[2 * d + 1];
  const float hd = lhead[d];
  const float c = cosf(hd), sn = sinf(hd);
  const float x = rx * c + ry * sn;
  const float y = -rx * sn + ry * c;
  F[i * 3 + 0] = sqrtf(x * x + y * y);
  F[i * 3 + 1] = atan2f(y, x);
  F[i * 3 + 2] = wrap_angle(chead[s] - hd);
}

__global__ void k_l2l_feat(const float* __restrict__ lpos, const float* __restrict__ lhead,
                           const int* __restrict__ src, const int* __restrict__ dst,
                           const int* __restrict__ type, const float* __restrict__ hop,
                           float* __restrict__ F, int E) {
  int i = blockIdx.x * blockDim.x + threadIdx.x;
  if (i >= E) return;
  const int s = src[i], d = dst[i];
  const float rx = lpos[2 * s] - lpos[2 * d];
  const float ry = lpos[2 * s + 1] - lpos[2 * d + 1];
  const float hd = lhead[d];
  const float c = cosf(hd), sn = sinf(hd);
  const float x = rx * c + ry * sn;
  const float y = -rx * sn + ry * c;
  const int ty = type[i];
  F[i * 7 + 0] = sqrtf(x * x + y * y);
  F[i * 7 + 1] = atan2f(y, x);
  F[i * 7 + 2] = wrap_angle(lhead[s] - hd);
  F[i * 7 + 3] = hop[i];
  F[i * 7 + 4] = (ty == 0) ? 1.0f : 0.0f;
  F[i * 7 + 5] = (ty == 1) ? 1.0f : 0.0f;
  F[i * 7 + 6] = (ty == 2) ? 1.0f : 0.0f;
}

// ------------------------------ attention core ------------------------------
// w[e,h] = dot16(q[dst,h], k[src,h] + ek[e,h]) / 4 ; wmax = segmax(w over dst)
__global__ void k_scores(const float* __restrict__ Q, const float* __restrict__ Kn,
                         const float* __restrict__ EK,
                         const int* __restrict__ src, const int* __restrict__ dst,
                         int E, float* __restrict__ w, unsigned* __restrict__ wmax) {
  const int lane = threadIdx.x & 31;
  const int wave = threadIdx.x >> 5;
  const int e = blockIdx.x * (blockDim.x >> 5) + wave;
  if (e >= E) return;
  const int s = src[e], d = dst[e];
  const float* q  = Q  + (size_t)d * HD_;
  const float* k  = Kn + (size_t)s * HD_;
  const float* ek = EK + (size_t)e * HD_;
  const int j0 = lane * 4;
  float acc = 0.0f;
#pragma unroll
  for (int t = 0; t < 4; ++t) acc += q[j0 + t] * (k[j0 + t] + ek[j0 + t]);
  acc += __shfl_xor(acc, 1, 32);
  acc += __shfl_xor(acc, 2, 32);   // lanes within a head group (4 lanes) agree
  acc *= 0.25f;                    // 1/sqrt(16)
  if ((lane & 3) == 0) {
    const int h = lane >> 2;
    w[(size_t)e * 8 + h] = acc;
    atomicMax(&wmax[(size_t)d * 8 + h], fenc(acc));
  }
}

// ew = exp(w - wmax[dst]) in place; den = segsum(ew)
__global__ void k_expsum(float* __restrict__ w, const unsigned* __restrict__ wmax,
                         float* __restrict__ den, const int* __restrict__ dst, int E) {
  int i = blockIdx.x * blockDim.x + threadIdx.x;
  if (i >= E * 8) return;
  const int e = i >> 3, h = i & 7;
  const int d = dst[e];
  const float m = fdec(wmax[(size_t)d * 8 + h]);
  const float v = expf(w[i] - m);
  w[i] = v;
  atomicAdd(&den[(size_t)d * 8 + h], v);
}

// AGG[dst,:] += (v[src,:] + ev[e,:]) * (ew[e,h]/den[dst,h])
__global__ void k_agg(const float* __restrict__ ew, const float* __restrict__ den,
                      const float* __restrict__ Vn, const float* __restrict__ EV,
                      const int* __restrict__ src, const int* __restrict__ dst,
                      int E, float* __restrict__ AGG) {
  const int lane = threadIdx.x & 31;
  const int wave = threadIdx.x >> 5;
  const int e = blockIdx.x * (blockDim.x >> 5) + wave;
  if (e >= E) return;
  const int s = src[e], d = dst[e];
  const int h = lane >> 2;
  const float a = ew[(size_t)e * 8 + h] / den[(size_t)d * 8 + h];
  const int j0 = lane * 4;
#pragma unroll
  for (int t = 0; t < 4; ++t) {
    const int j = j0 + t;
    atomicAdd(&AGG[(size_t)d * HD_ + j],
              (Vn[(size_t)s * HD_ + j] + EV[(size_t)e * HD_ + j]) * a);
  }
}

__global__ void k_add(const float* __restrict__ A, const float* __restrict__ B,
                      float* __restrict__ O, size_t n) {
  size_t i = (size_t)blockIdx.x * blockDim.x + threadIdx.x;
  if (i < n) O[i] = A[i] + B[i];
}

// --------------------------------- host side --------------------------------

struct AttnP {
  const float *Wek, *Wev, *Wk, *Wq, *Wv;
  const float *bek, *bev, *bk, *bq, *bv;
  const float *fW1, *fW2, *fb1, *fb2;
  const float *ln_dst_b, *ln_dst_g;
  const float *ln_e_b, *ln_e_g, *ln_f_b, *ln_f_g, *ln_src_b, *ln_src_g;
};
struct AttnW {  // f16-transposed weights
  _Float16 *qt, *kt, *vt, *ekt, *evt, *f1t, *f2t;
};
struct EmbP {
  const float *W1, *W2, *b1, *b2, *bln, *g;
};

static inline size_t cdiv(size_t a, size_t b) { return (a + b - 1) / b; }

static void launch_gemm(hipStream_t st, const _Float16* A, const _Float16* Bt,
                        const float* bias, float* Cf, _Float16* Ch,
                        int M, int N, int K, int relu) {
  dim3 grid((unsigned)cdiv(M, 16 * GEMM_WAVES), (unsigned)(N / (GEMM_NT * 16)));
  if (K == 128) {
    k_wmma_gemm_t<128><<<grid, 32 * GEMM_WAVES, 0, st>>>(A, Bt, bias, Cf, Ch, M, N, relu);
  } else {
    k_wmma_gemm_t<512><<<grid, 32 * GEMM_WAVES, 0, st>>>(A, Bt, bias, Cf, Ch, M, N, relu);
  }
}

static void conv_w(hipStream_t st, const float* W, _Float16* Bt, int K, int N) {
  k_w_to_f16t<<<(unsigned)cdiv((size_t)K * N, 256), 256, 0, st>>>(W, Bt, K, N);
}

extern "C" void kernel_launch(void* const* d_in, const int* in_sizes, int n_in,
                              void* d_out, int out_size, void* d_ws, size_t ws_size,
                              hipStream_t stream) {
  (void)in_sizes; (void)n_in; (void)out_size; (void)ws_size;

  // ---- inputs (setup_inputs dict order; params pytree in sorted-key order) ----
  const float* c_len  = (const float*)d_in[0];
  const float* c_pos  = (const float*)d_in[1];
  const float* c_head = (const float*)d_in[2];
  const float* l_len  = (const float*)d_in[3];
  const float* l_pos  = (const float*)d_in[4];
  const float* l_head = (const float*)d_in[5];
  const float* l_int  = (const float*)d_in[6];
  const float* l_turn = (const float*)d_in[7];
  const float* l_tc   = (const float*)d_in[8];
  const int* c2l_src  = (const int*)d_in[9];
  const int* c2l_dst  = (const int*)d_in[10];
  const int* l2l_src  = (const int*)d_in[11];
  const int* l2l_dst  = (const int*)d_in[12];
  const int* l2l_type = (const int*)d_in[13];
  const float* l2l_hop = (const float*)d_in[14];

  const int P0 = 15;
  const float* const* pp = (const float* const*)(d_in + P0);

  // c2l_attn leaves (sorted keys: W* < b* < ffn < ln_*)
  AttnP c2l;
  c2l.Wek = pp[0];  c2l.Wev = pp[1];  c2l.Wk = pp[2];  c2l.Wq = pp[3];  c2l.Wv = pp[4];
  c2l.bek = pp[5];  c2l.bev = pp[6];  c2l.bk = pp[7];  c2l.bq = pp[8];  c2l.bv = pp[9];
  c2l.fW1 = pp[10]; c2l.fW2 = pp[11]; c2l.fb1 = pp[12]; c2l.fb2 = pp[13];
  c2l.ln_dst_b = pp[14]; c2l.ln_dst_g = pp[15];
  c2l.ln_e_b = pp[16]; c2l.ln_e_g = pp[17];
  c2l.ln_f_b = pp[18]; c2l.ln_f_g = pp[19];
  c2l.ln_src_b = pp[20]; c2l.ln_src_g = pp[21];
  EmbP c2l_emb = { pp[22], pp[23], pp[24], pp[25], pp[26], pp[27] };
  EmbP c_emb   = { pp[28], pp[29], pp[30], pp[31], pp[32], pp[33] };
  AttnP l2l;
  l2l.Wek = pp[34]; l2l.Wev = pp[35]; l2l.Wk = pp[36]; l2l.Wq = pp[37]; l2l.Wv = pp[38];
  l2l.bek = pp[39]; l2l.bev = pp[40]; l2l.bk = pp[41]; l2l.bq = pp[42]; l2l.bv = pp[43];
  l2l.fW1 = pp[44]; l2l.fW2 = pp[45]; l2l.fb1 = pp[46]; l2l.fb2 = pp[47];
  l2l.ln_dst_b = nullptr; l2l.ln_dst_g = nullptr;
  l2l.ln_e_b = pp[48]; l2l.ln_e_g = pp[49];
  l2l.ln_f_b = pp[50]; l2l.ln_f_g = pp[51];
  l2l.ln_src_b = pp[52]; l2l.ln_src_g = pp[53];
  EmbP l2l_emb = { pp[54], pp[55], pp[56], pp[57], pp[58], pp[59] };
  EmbP l_emb   = { pp[60], pp[61], pp[62], pp[63], pp[64], pp[65] };

  // ---- workspace bump allocator ----
  char* base = (char*)d_ws;
  size_t off = 0;
  auto alloc = [&](size_t bytes) -> void* {
    void* p = base + off;
    off = (off + bytes + 255) & ~(size_t)255;
    return p;
  };
  const size_t EMAX = EL_;  // >= EC_

  // f16 transposed weights
  AttnW wc, wl;
  _Float16* embW2t[4];
  auto alloc_attn_w = [&](AttnW& w) {
    w.qt  = (_Float16*)alloc(128 * 128 * 2);
    w.kt  = (_Float16*)alloc(128 * 128 * 2);
    w.vt  = (_Float16*)alloc(128 * 128 * 2);
    w.ekt = (_Float16*)alloc(128 * 128 * 2);
    w.evt = (_Float16*)alloc(128 * 128 * 2);
    w.f1t = (_Float16*)alloc(128 * 512 * 2);
    w.f2t = (_Float16*)alloc(512 * 128 * 2);
  };
  alloc_attn_w(wc);
  alloc_attn_w(wl);
  for (int i = 0; i < 4; ++i) embW2t[i] = (_Float16*)alloc(128 * 128 * 2);

  _Float16* E16   = (_Float16*)alloc(EMAX * HD_ * 2);      // edge h / e_ln (f16)
  float*    EF32A = (float*)alloc(EMAX * HD_ * 4);         // e f32, then EK
  float*    EF32B = (float*)alloc(EMAX * HD_ * 4);         // EV
  float*    EFEAT = (float*)alloc(EMAX * 8 * 4);           // raw edge features
  float*    W8    = (float*)alloc(EMAX * 8 * 4);           // scores / ew
  unsigned* WMAX  = (unsigned*)alloc(NL_ * 8 * 4);
  float*    DEN   = (float*)alloc(NL_ * 8 * 4);
  _Float16* N16A  = (_Float16*)alloc((size_t)NC_ * HD_ * 2); // xs f16 / c-front h
  _Float16* N16B  = (_Float16*)alloc((size_t)NL_ * HD_ * 2); // xd f16 / of16 / l-front h
  float*    CEMB  = (float*)alloc((size_t)NC_ * HD_ * 4);
  float*    LEMB  = (float*)alloc((size_t)NL_ * HD_ * 4);
  float*    XD    = (float*)alloc((size_t)NL_ * HD_ * 4);
  float*    QN    = (float*)alloc((size_t)NL_ * HD_ * 4);
  float*    KN    = (float*)alloc((size_t)NC_ * HD_ * 4);
  float*    VN    = (float*)alloc((size_t)NC_ * HD_ * 4);
  float*    AGG   = (float*)alloc((size_t)NL_ * HD_ * 4);
  float*    OBUF  = (float*)alloc((size_t)NL_ * HD_ * 4);
  _Float16* F1    = (_Float16*)alloc((size_t)NL_ * 512 * 2);
  float*    F2    = (float*)alloc((size_t)NL_ * HD_ * 4);

  // ---- phase 0: weight conversion (f32 -> f16 transposed) ----
  conv_w(stream, c2l.Wq, wc.qt, 128, 128);  conv_w(stream, c2l.Wk, wc.kt, 128, 128);
  conv_w(stream, c2l.Wv, wc.vt, 128, 128);  conv_w(stream, c2l.Wek, wc.ekt, 128, 128);
  conv_w(stream, c2l.Wev, wc.evt, 128, 128);
  conv_w(stream, c2l.fW1, wc.f1t, 128, 512); conv_w(stream, c2l.fW2, wc.f2t, 512, 128);
  conv_w(stream, l2l.Wq, wl.qt, 128, 128);  conv_w(stream, l2l.Wk, wl.kt, 128, 128);
  conv_w(stream, l2l.Wv, wl.vt, 128, 128);  conv_w(stream, l2l.Wek, wl.ekt, 128, 128);
  conv_w(stream, l2l.Wev, wl.evt, 128, 128);
  conv_w(stream, l2l.fW1, wl.f1t, 128, 512); conv_w(stream, l2l.fW2, wl.f2t, 512, 128);
  conv_w(stream, c_emb.W2, embW2t[0], 128, 128);
  conv_w(stream, l_emb.W2, embW2t[1], 128, 128);
  conv_w(stream, c2l_emb.W2, embW2t[2], 128, 128);
  conv_w(stream, l2l_emb.W2, embW2t[3], 128, 128);

  // ---- phase 1: node embeddings (mlp2) ----
  k_front<<<(unsigned)cdiv(NC_, 8), 256, 0, stream>>>(c_len, 1, NC_,
      c_emb.W1, c_emb.b1, c_emb.g, c_emb.bln, N16A);
  launch_gemm(stream, N16A, embW2t[0], c_emb.b2, CEMB, nullptr, NC_, 128, 128, 0);

  k_pack_lfeat<<<(unsigned)cdiv(NL_, 256), 256, 0, stream>>>(l_len, l_int, l_turn, l_tc,
                                                             EFEAT, NL_);
  k_front<<<(unsigned)cdiv(NL_, 8), 256, 0, stream>>>(EFEAT, 4, NL_,
      l_emb.W1, l_emb.b1, l_emb.g, l_emb.bln, N16B);
  launch_gemm(stream, N16B, embW2t[1], l_emb.b2, LEMB, nullptr, NL_, 128, 128, 0);

  // ---- generic attention layer ----
  auto attn_layer = [&](const AttnP& p, const AttnW& w, const EmbP& ep,
                        const _Float16* emb_W2t, int din,
                        int E, const int* src, const int* dst,
                        const float* xsrc_f32, int Ns,
                        const float* xdst_f32, bool self_attn, float* out_final) {
    // edge embedding: feat -> front -> W2 -> LN_e (f16)
    k_front<<<(unsigned)cdiv(E, 8), 256, 0, stream>>>(EFEAT, din, E,
        ep.W1, ep.b1, ep.g, ep.bln, E16);
    launch_gemm(stream, E16, emb_W2t, ep.b2, EF32A, nullptr, E, 128, 128, 0);
    k_ln128<<<(unsigned)cdiv(E, 8), 256, 0, stream>>>(EF32A, p.ln_e_g, p.ln_e_b, E,
                                                      nullptr, E16);
    // node LNs
    if (self_attn) {
      k_ln128<<<(unsigned)cdiv(Ns, 8), 256, 0, stream>>>(xsrc_f32, p.ln_src_g, p.ln_src_b,
                                                         Ns, XD, N16A);
    } else {
      k_ln128<<<(unsigned)cdiv(Ns, 8), 256, 0, stream>>>(xsrc_f32, p.ln_src_g, p.ln_src_b,
                                                         Ns, nullptr, N16A);
      k_ln128<<<(unsigned)cdiv(NL_, 8), 256, 0, stream>>>(xdst_f32, p.ln_dst_g, p.ln_dst_b,
                                                          NL_, XD, N16B);
    }
    const _Float16* xd16 = self_attn ? N16A : N16B;
    // projections (WMMA)
    launch_gemm(stream, xd16, w.qt, p.bq, QN, nullptr, NL_, 128, 128, 0);
    launch_gemm(stream, N16A, w.kt, p.bk, KN, nullptr, Ns, 128, 128, 0);
    launch_gemm(stream, N16A, w.vt, p.bv, VN, nullptr, Ns, 128, 128, 0);
    launch_gemm(stream, E16, w.ekt, p.bek, EF32A, nullptr, E, 128, 128, 0);
    launch_gemm(stream, E16, w.evt, p.bev, EF32B, nullptr, E, 128, 128, 0);
    // segment softmax + aggregate
    hipMemsetAsync(WMAX, 0, (size_t)NL_ * 8 * 4, stream);   // fenc order: 0 == lowest
    hipMemsetAsync(DEN, 0, (size_t)NL_ * 8 * 4, stream);
    hipMemsetAsync(AGG, 0, (size_t)NL_ * HD_ * 4, stream);
    k_scores<<<(unsigned)cdiv(E, 8), 256, 0, stream>>>(QN, KN, EF32A, src, dst, E, W8, WMAX);
    k_expsum<<<(unsigned)cdiv((size_t)E * 8, 256), 256, 0, stream>>>(W8, WMAX, DEN, dst, E);
    k_agg<<<(unsigned)cdiv(E, 8), 256, 0, stream>>>(W8, DEN, VN, EF32B, src, dst, E, AGG);
    // out = xd + agg ; final = out + ffn(LN(out))
    k_add<<<(unsigned)cdiv((size_t)NL_ * HD_, 256), 256, 0, stream>>>(XD, AGG, OBUF,
                                                                      (size_t)NL_ * HD_);
    k_ln128<<<(unsigned)cdiv(NL_, 8), 256, 0, stream>>>(OBUF, p.ln_f_g, p.ln_f_b, NL_,
                                                        nullptr, N16B);
    launch_gemm(stream, N16B, w.f1t, p.fb1, nullptr, F1, NL_, 512, 128, 1);
    launch_gemm(stream, F1, w.f2t, p.fb2, F2, nullptr, NL_, 128, 512, 0);
    k_add<<<(unsigned)cdiv((size_t)NL_ * HD_, 256), 256, 0, stream>>>(OBUF, F2, out_final,
                                                                      (size_t)NL_ * HD_);
  };

  // ---- phase 2+3: c2l cross attention ----
  k_c2l_feat<<<(unsigned)cdiv(EC_, 256), 256, 0, stream>>>(c_pos, l_pos, c_head, l_head,
                                                           c2l_src, c2l_dst, EFEAT, EC_);
  attn_layer(c2l, wc, c2l_emb, embW2t[2], 3, EC_, c2l_src, c2l_dst,
             CEMB, NC_, LEMB, /*self=*/false, LEMB);

  // ---- phase 4+5: l2l self attention ----
  k_l2l_feat<<<(unsigned)cdiv(EL_, 256), 256, 0, stream>>>(l_pos, l_head, l2l_src, l2l_dst,
                                                           l2l_type, l2l_hop, EFEAT, EL_);
  attn_layer(l2l, wl, l2l_emb, embW2t[3], 7, EL_, l2l_src, l2l_dst,
             LEMB, NL_, LEMB, /*self=*/true, (float*)d_out);
}
